// SparseTopKAttention_32280974197468
// MI455X (gfx1250) — matrix operand order repaired and checked by reading, not compiled, not run on hardware
//
#include <hip/hip_runtime.h>
#include <stdint.h>

// Problem constants (match reference)
#define BB 2
#define SS 2048
#define DD 1024
#define HH 16
#define HD 64
#define TOPK 32

typedef unsigned short u16;
typedef __attribute__((ext_vector_type(16))) __bf16 v16bf;
typedef __attribute__((ext_vector_type(8)))  float  v8f;

union BF16Frag {
    v16bf v;
    uint4 qv[2];
    unsigned int u[8];
};

__device__ __forceinline__ v8f wmma_bf16(const BF16Frag& a, const BF16Frag& b, v8f c) {
    // D = A(16x32 bf16) * B(32x16 bf16) + C(16x16 f32)
    return __builtin_amdgcn_wmma_f32_16x16x32_bf16(false, a.v, false, b.v, (short)0, c, false, false);
}

__device__ __forceinline__ u16 f2bf(float x) {           // RNE f32 -> bf16
    uint32_t u = __float_as_uint(x);
    uint32_t r = (u + 0x7FFFu + ((u >> 16) & 1u)) >> 16;
    return (u16)r;
}
__device__ __forceinline__ float bf2f(u16 h) {
    return __uint_as_float(((uint32_t)h) << 16);
}
// order-preserving f32 <-> u32 transform (total order, -0 < +0)
__device__ __forceinline__ unsigned ftrans(float f) {
    unsigned u = __float_as_uint(f);
    return (u & 0x80000000u) ? ~u : (u | 0x80000000u);
}
__device__ __forceinline__ float funtrans(unsigned t) {
    unsigned v = (t & 0x80000000u) ? (t ^ 0x80000000u) : ~t;
    return __uint_as_float(v);
}

__device__ __forceinline__ int wsum_i(int v) {
    for (int o = 16; o; o >>= 1) v += __shfl_xor(v, o, 32);
    return v;
}
__device__ __forceinline__ float wsum_f(float v) {
    for (int o = 16; o; o >>= 1) v += __shfl_xor(v, o, 32);
    return v;
}
__device__ __forceinline__ unsigned wmax_u(unsigned v) {
    for (int o = 16; o; o >>= 1) {
        unsigned t = (unsigned)__shfl_xor((int)v, o, 32);
        v = (t > v) ? t : v;
    }
    return v;
}

// ---------------------------------------------------------------------------
// Generic WMMA GEMM: C(4096 x NT) = A(4096 x 1024) * W(1024 x NT) + bias
// Block: 256 thr (8 waves), tile 64(M) x 64(N); wave tile 16 x 32 (2 C frags).
// MODE 0: scatter to q/k/v bf16 buffers in (B,H,S,HD) layout (q scaled 1/8)
// MODE 1: non-temporal f32 row-major store to outf (write-once stream)
// ---------------------------------------------------------------------------
#define SAP 40   // LDS row stride (ushorts): 80B rows -> uint4-aligned

template<int NT, int MODE>
__global__ __launch_bounds__(256)
void gemm_wmma(const float* __restrict__ A, const float* __restrict__ W,
               const float* __restrict__ bias,
               u16* __restrict__ qd, u16* __restrict__ kd, u16* __restrict__ vd,
               float* __restrict__ outf)
{
    __shared__ __align__(16) u16 sA[64][SAP];   // [m][k]
    __shared__ __align__(16) u16 sB[64][SAP];   // [n][k]  (W tile transposed)

    const int m0 = blockIdx.y * 64;
    const int n0 = blockIdx.x * 64;
    const int tid  = threadIdx.x;
    const int lane = tid & 31;
    const int wv   = tid >> 5;
    const int waveM = wv & 3;       // 0..3  -> 16-row slice
    const int waveN = wv >> 2;      // 0..1  -> 32-col slice
    const int half = lane >> 4;
    const int mr   = lane & 15;

    v8f c0 = {}; v8f c1 = {};

    for (int k0 = 0; k0 < 1024; k0 += 32) {
        // stage A tile 64x32 (f32 -> bf16)
        {
            int k = tid & 31, m = tid >> 5;            // m 0..7
            #pragma unroll
            for (int i = 0; i < 8; ++i)
                sA[m + 8 * i][k] = f2bf(A[(size_t)(m0 + m + 8 * i) * 1024 + k0 + k]);
        }
        // stage W tile 32x64 transposed -> sB[n][k]
        {
            int n = tid & 63, kk = tid >> 6;           // kk 0..3
            #pragma unroll
            for (int i = 0; i < 8; ++i)
                sB[n][kk + 4 * i] = f2bf(W[(size_t)(k0 + kk + 4 * i) * NT + n0 + n]);
        }
        __syncthreads();

        // A frag: 16-bit A 16x32 layout. lane(0-15): regs0-3 K=0..7, regs4-7 K=16..23;
        // lane(16-31): +8.  -> two uint4 loads per lane.
        BF16Frag a, b0, b1;
        const u16* ar  = &sA[waveM * 16 + mr][0];
        a.qv[0] = *(const uint4*)(ar + 8 * half);
        a.qv[1] = *(const uint4*)(ar + 8 * half + 16);
        // B frag: 16-bit B 32x16. n = lane%16, same K order as A:
        // regs0-3 K=8*(lane/16)..+7, regs4-7 K=16+8*(lane/16)..+7 from sB[n][.]
        const u16* br0 = &sB[waveN * 32 + mr][0];
        const u16* br1 = &sB[waveN * 32 + 16 + mr][0];
        b0.qv[0] = *(const uint4*)(br0 + 8 * half);
        b0.qv[1] = *(const uint4*)(br0 + 8 * half + 16);
        b1.qv[0] = *(const uint4*)(br1 + 8 * half);
        b1.qv[1] = *(const uint4*)(br1 + 8 * half + 16);

        c0 = wmma_bf16(a, b0, c0);
        c1 = wmma_bf16(a, b1, c1);
        __syncthreads();
    }

    // store: C layout -> reg r: M = r + 8*half, N = lane%16
    #pragma unroll
    for (int ct = 0; ct < 2; ++ct) {
        v8f& c = ct ? c1 : c0;
        #pragma unroll
        for (int r = 0; r < 8; ++r) {
            int gm = m0 + waveM * 16 + r + 8 * half;
            int gn = n0 + waveN * 32 + ct * 16 + mr;
            float val = c[r] + bias[gn];
            if (MODE == 0) {
                int t = gn / DD, rem = gn % DD;
                int h = rem / HD, d = rem % HD;
                int b = gm / SS, s = gm % SS;
                if (t == 0) val *= 0.125f;                 // fold 1/sqrt(HD) into q
                u16* dst = (t == 0) ? qd : (t == 1) ? kd : vd;
                dst[(((size_t)b * HH + h) * SS + s) * HD + d] = f2bf(val);
            } else {
                // write-once final output: non-temporal (TH=NT) store
                __builtin_nontemporal_store(val, &outf[(size_t)gm * NT + gn]);
            }
        }
    }
}

// ---------------------------------------------------------------------------
// Attention: one block per (b, h, 16-query tile). 256 thr = 8 waves.
// Phase 1: WMMA scores (bf16 q/k straight from global) -> 16x2048 f32 in LDS.
// Phase 2: per row (2 rows/wave): exact radix-select of 32nd largest, masked
//          softmax, dense attn row NT-stored to d_out, ballot-compacted list.
// Phase 3: sparse f32 attn @ v -> context buffer.
// ---------------------------------------------------------------------------
#define SSTR 2056          // score row stride (floats)
#define CAP  64            // kept-index capacity per row (ties safety margin)
#define ATTN_LDS_BYTES (16 * SSTR * 4 + 16 * 4 + 16 * CAP * 4 + 16 * CAP * 4)

__global__ __launch_bounds__(256)
void attn_topk(const u16* __restrict__ qb, const u16* __restrict__ kb,
               const u16* __restrict__ vb,
               float* __restrict__ attnOut, float* __restrict__ ctx)
{
    extern __shared__ char smem[];
    float* ssc    = (float*)smem;                                 // [16][SSTR]
    int*   s_cnt  = (int*)  (smem + 16 * SSTR * 4);               // [16]
    int*   s_idx  = s_cnt + 16;                                   // [16][CAP]
    float* s_prob = (float*)(s_idx + 16 * CAP);                   // [16][CAP]

    const int qt = blockIdx.x, h = blockIdx.y, b = blockIdx.z;
    const int bh = b * HH + h;
    const int q0 = qt * 16;
    const int tid  = threadIdx.x;
    const int lane = tid & 31;
    const int wv   = tid >> 5;
    const int half = lane >> 4;
    const int mr   = lane & 15;

    // ---- Phase 1: scores. Wave wv covers keys [wv*256, wv*256+256). ----
    const u16* qrow = qb + (((size_t)bh * SS) + q0 + mr) * HD;
    BF16Frag a0, a1;                       // K-steps 0..31 and 32..63
    a0.qv[0] = *(const uint4*)(qrow + 8 * half);
    a0.qv[1] = *(const uint4*)(qrow + 8 * half + 16);
    a1.qv[0] = *(const uint4*)(qrow + 8 * half + 32);
    a1.qv[1] = *(const uint4*)(qrow + 8 * half + 48);

    for (int kt = 0; kt < 16; ++kt) {
        int keyBase = wv * 256 + kt * 16;
        const u16* krow = kb + (((size_t)bh * SS) + keyBase + mr) * HD;
        BF16Frag f0, f1;
        f0.qv[0] = *(const uint4*)(krow + 8 * half);
        f0.qv[1] = *(const uint4*)(krow + 8 * half + 16);
        f1.qv[0] = *(const uint4*)(krow + 8 * half + 32);
        f1.qv[1] = *(const uint4*)(krow + 8 * half + 48);
        v8f c = {};
        c = wmma_bf16(a0, f0, c);
        c = wmma_bf16(a1, f1, c);
        #pragma unroll
        for (int r = 0; r < 8; ++r)
            ssc[(size_t)(r + 8 * half) * SSTR + keyBase + mr] = c[r];
    }
    __syncthreads();

    // ---- Phase 2: rows 2*wv and 2*wv+1 ----
    for (int rr = 0; rr < 2; ++rr) {
        int row = wv * 2 + rr;
        float* srow = ssc + (size_t)row * SSTR;

        unsigned ui[64];
        #pragma unroll
        for (int i = 0; i < 64; ++i) ui[i] = ftrans(srow[lane + 32 * i]);

        unsigned um = 0;
        #pragma unroll
        for (int i = 0; i < 64; ++i) um = (ui[i] > um) ? ui[i] : um;
        um = wmax_u(um);

        // radix binary search: largest t with count(x >= t) >= TOPK == 32nd largest
        unsigned cur = 0;
        for (int bit = 31; bit >= 0; --bit) {
            unsigned cand = cur | (1u << bit);
            int cnt = 0;
            #pragma unroll
            for (int i = 0; i < 64; ++i) cnt += (ui[i] >= cand) ? 1 : 0;
            cnt = wsum_i(cnt);
            if (cnt >= TOPK) cur = cand;
        }

        float fmaxv = funtrans(um);
        float lsum = 0.f;
        #pragma unroll
        for (int i = 0; i < 64; ++i) {
            float e = 0.f;
            if (ui[i] >= cur) e = __expf(funtrans(ui[i]) - fmaxv);
            srow[lane + 32 * i] = e;
            lsum += e;
        }
        float inv = 1.0f / wsum_f(lsum);

        // 512 MB write-once stream: non-temporal stores keep keys/values in L2
        float* arow = attnOut + (((size_t)bh * SS) + q0 + row) * SS;
        int base = 0;
        #pragma unroll
        for (int i = 0; i < 64; ++i) {
            int j = lane + 32 * i;
            float p = srow[j] * inv;
            __builtin_nontemporal_store(p, &arow[j]);      // dense attn output
            bool keep = (ui[i] >= cur);
            unsigned m = (unsigned)__ballot(keep);         // wave32: low 32 bits
            int pos = base + __popc(m & ((1u << lane) - 1u));
            if (keep && pos < CAP) {
                s_idx[row * CAP + pos]  = j;
                s_prob[row * CAP + pos] = p;
            }
            base += __popc(m);
        }
        if (lane == 0) s_cnt[row] = (base < CAP) ? base : CAP;
    }
    __syncthreads();

    // ---- Phase 3: sparse attn @ v (f32). lane owns dims lane, lane+32 ----
    for (int rr = 0; rr < 2; ++rr) {
        int row = wv * 2 + rr;
        int cnt = s_cnt[row];
        float acc0 = 0.f, acc1 = 0.f;
        for (int i = 0; i < cnt; ++i) {
            int   j = s_idx[row * CAP + i];
            float w = s_prob[row * CAP + i];
            const u16* vrow = vb + (((size_t)bh * SS) + j) * HD;
            acc0 += w * bf2f(vrow[lane]);
            acc1 += w * bf2f(vrow[lane + 32]);
        }
        // ctx is re-read by the projection GEMM: keep RT (L2-resident, 16 MB)
        float* crow = ctx + ((size_t)b * SS + q0 + row) * DD + h * HD;
        crow[lane]      = acc0;
        crow[lane + 32] = acc1;
    }
}

// ---------------------------------------------------------------------------
extern "C" void kernel_launch(void* const* d_in, const int* in_sizes, int n_in,
                              void* d_out, int out_size, void* d_ws, size_t ws_size,
                              hipStream_t stream)
{
    (void)in_sizes; (void)n_in; (void)out_size; (void)ws_size;
    const float* query  = (const float*)d_in[0];
    // d_in[1] (key) and d_in[2] (value) are unused by the reference.
    const float* qkv_w  = (const float*)d_in[3];
    const float* qkv_b  = (const float*)d_in[4];
    const float* proj_w = (const float*)d_in[5];
    const float* proj_b = (const float*)d_in[6];

    float* out     = (float*)d_out;                       // (B,S,D)
    float* attnOut = out + (size_t)BB * SS * DD;          // (B,H,S,S)

    const size_t NHS = (size_t)BB * HH * SS * HD;         // 4M elements
    u16*   qbuf = (u16*)d_ws;
    u16*   kbuf = qbuf + NHS;
    u16*   vbuf = kbuf + NHS;
    float* ctx  = (float*)(vbuf + NHS);                   // (B,S,D) f32

    // large dynamic LDS (~137 KB) — CDNA5 WGP supports 320 KB
    (void)hipFuncSetAttribute((const void*)attn_topk,
                              hipFuncAttributeMaxDynamicSharedMemorySize,
                              ATTN_LDS_BYTES);

    dim3 blk(256);
    gemm_wmma<3072, 0><<<dim3(48, 64), blk, 0, stream>>>(
        query, qkv_w, qkv_b, qbuf, kbuf, vbuf, nullptr);

    attn_topk<<<dim3(SS / 16, HH, BB), blk, ATTN_LDS_BYTES, stream>>>(
        qbuf, kbuf, vbuf, attnOut, ctx);

    gemm_wmma<1024, 1><<<dim3(16, 64), blk, 0, stream>>>(
        ctx, proj_w, proj_b, nullptr, nullptr, nullptr, out);
}
